// CausalSelfAttention_8890582302957
// MI455X (gfx1250) — matrix operand order repaired and checked by reading, not compiled, not run on hardware
//
#include <hip/hip_runtime.h>
#include <hip/hip_bf16.h>

typedef unsigned short u16;
typedef __attribute__((ext_vector_type(16))) __bf16 v16bf;
typedef __attribute__((ext_vector_type(8)))  float  v8f;

union Frag {           // one WMMA 16-bit A/B operand: 16 bf16 = 32 bytes = 8 VGPRs
    uint4 q[2];
    v16bf v;
};

__device__ __forceinline__ u16 f32_to_bf16(float f) {
    union { float f; unsigned int u; } cv; cv.f = f;
    unsigned int u = cv.u;
    u += 0x7FFFu + ((u >> 16) & 1u);   // round to nearest even
    return (u16)(u >> 16);
}

// ---------------------------------------------------------------------------
// CDNA5 async global->LDS copy (ASYNCcnt-tracked), 16 bytes per lane.
// Guarded: falls back to a synchronous copy if the builtin is unavailable.
// ---------------------------------------------------------------------------
#if defined(__gfx1250__) && __has_builtin(__builtin_amdgcn_global_load_async_to_lds_b128)
#define HAS_ASYNC_LDS 1
#else
#define HAS_ASYNC_LDS 0
#endif

#if HAS_ASYNC_LDS
typedef int v4i_b128 __attribute__((vector_size(16)));
typedef __attribute__((address_space(1))) v4i_b128* g_b128_t;
typedef __attribute__((address_space(3))) v4i_b128* l_b128_t;
#endif

__device__ __forceinline__ void cp16(void* lds, const void* g) {
#if HAS_ASYNC_LDS
    __builtin_amdgcn_global_load_async_to_lds_b128(
        (g_b128_t)(unsigned long long)g,
        (l_b128_t)(unsigned int)(unsigned long long)lds,
        0, 0);
#else
    *(uint4*)lds = *(const uint4*)g;
#endif
}

__device__ __forceinline__ void async_wait() {
#if HAS_ASYNC_LDS
#if __has_builtin(__builtin_amdgcn_s_wait_asynccnt)
    __builtin_amdgcn_s_wait_asynccnt(0);
#else
    asm volatile("s_wait_asynccnt 0" ::: "memory");
#endif
#endif
}

// ---------------------------------------------------------------------------
// fp32 -> bf16 conversion (x)
// ---------------------------------------------------------------------------
__global__ __launch_bounds__(256) void cvt_bf16(const float* __restrict__ in,
                                                u16* __restrict__ out, int n) {
    int i = blockIdx.x * blockDim.x + threadIdx.x;
    int stride = gridDim.x * blockDim.x;
    for (; i < n; i += stride) out[i] = f32_to_bf16(in[i]);
}

// ---------------------------------------------------------------------------
// fp32 W(K,N) -> bf16 Wt(N,K): one-time tiled transpose (64x64 tiles via LDS)
// so GEMM B-tiles become contiguous 16B copies (async-LDS friendly).
// ---------------------------------------------------------------------------
__global__ __launch_bounds__(256) void cvt_w_transpose(const float* __restrict__ W,
                                                       u16* __restrict__ Wt) {
    __shared__ float tile[64][65];
    const int t  = threadIdx.x;
    const int r0 = blockIdx.y * 64, c0 = blockIdx.x * 64;
#pragma unroll
    for (int i = 0; i < 16; ++i) {
        int idx = t + 256 * i;
        int rr = idx >> 6, cc = idx & 63;
        tile[rr][cc] = W[(size_t)(r0 + rr) * 1024 + c0 + cc];
    }
    __syncthreads();
#pragma unroll
    for (int i = 0; i < 16; ++i) {
        int idx = t + 256 * i;
        int rr = idx >> 6, cc = idx & 63;
        Wt[(size_t)(c0 + rr) * 1024 + r0 + cc] = f32_to_bf16(tile[cc][rr]);
    }
}

// ---------------------------------------------------------------------------
// bf16 GEMM:  C(M=8192,N=1024) = A(M,K=1024) * Wt(N,K)^T + bias
// block tile 128x128, 256 threads = 8 waves (4x2), wave tile 32x64 (2x4 accs).
// Double-buffered LDS; next k-tile prefetched via async-to-LDS during WMMAs.
// mode 0: bf16 out, (B,H,S,HD) layout        (Q,K projections)
// mode 2: bf16 out, (B,H,HD,S) layout        (V projection, pre-transposed)
// mode 1: fp32 out, row-major                (final projection)
// ---------------------------------------------------------------------------
__global__ __launch_bounds__(256) void gemm_bf16(const u16* __restrict__ A,
                                                 const u16* __restrict__ Wt,
                                                 const float* __restrict__ bias,
                                                 u16* __restrict__ outq,
                                                 float* __restrict__ outf,
                                                 int mode) {
    constexpr int K = 1024, N = 1024;
    __shared__ __align__(16) u16 As[2][128][32];   // [buf][m][k]
    __shared__ __align__(16) u16 Bs[2][128][32];   // [buf][n][k]

    const int t    = threadIdx.x;
    const int lane = t & 31;
    const int wave = t >> 5;
    const int wm   = wave >> 1;          // 0..3  (32-row group)
    const int wn   = wave & 1;           // 0..1  (64-col group)
    const int lh   = lane >> 4;          // lane half
    const int lr   = lane & 15;
    const int m0   = blockIdx.y * 128;
    const int n0   = blockIdx.x * 128;

    auto stage = [&](int buf, int kb) {
#pragma unroll
        for (int i = 0; i < 2; ++i) {
            int c   = t + 256 * i;             // 0..511
            int row = c >> 2;
            int ck  = (c & 3) << 3;
            cp16(&As[buf][row][ck], &A [(size_t)(m0 + row) * K + kb + ck]);
            cp16(&Bs[buf][row][ck], &Wt[(size_t)(n0 + row) * K + kb + ck]);
        }
    };

    v8f acc[2][4] = {};
    stage(0, 0);

    int buf = 0;
    for (int kb = 0; kb < K; kb += 32, buf ^= 1) {
        async_wait();
        __syncthreads();                       // tiles[buf] ready; other buf free
        if (kb + 32 < K) stage(buf ^ 1, kb + 32);

        Frag a[2];
#pragma unroll
        for (int mt = 0; mt < 2; ++mt) {
            int row = wm * 32 + mt * 16 + lr;
            a[mt].q[0] = *(const uint4*)&As[buf][row][lh * 8];
            a[mt].q[1] = *(const uint4*)&As[buf][row][16 + lh * 8];
        }
#pragma unroll
        for (int nt = 0; nt < 4; ++nt) {
            Frag b;
            int col = wn * 64 + nt * 16 + lr;
            b.q[0] = *(const uint4*)&Bs[buf][col][lh * 8];
            b.q[1] = *(const uint4*)&Bs[buf][col][16 + lh * 8];
#pragma unroll
            for (int mt = 0; mt < 2; ++mt) {
                acc[mt][nt] = __builtin_amdgcn_wmma_f32_16x16x32_bf16(
                    false, a[mt].v, false, b.v, (short)0, acc[mt][nt], false, false);
            }
        }
    }

    // store: C layout -> lane lr is column, VGPR r is row (r + 8*lh)
#pragma unroll
    for (int mt = 0; mt < 2; ++mt) {
#pragma unroll
        for (int nt = 0; nt < 4; ++nt) {
#pragma unroll
            for (int r = 0; r < 8; ++r) {
                int row = m0 + wm * 32 + mt * 16 + r + 8 * lh;   // b*S + s
                int col = n0 + wn * 64 + nt * 16 + lr;           // h*64 + hd
                float val = acc[mt][nt][r] + bias[col];
                if (mode == 0) {                                 // (b,h,s,hd)
                    int bb = row >> 11, s = row & 2047;
                    int h  = col >> 6,  hd = col & 63;
                    outq[((size_t)(bb * 16 + h) * 2048 + s) * 64 + hd] =
                        f32_to_bf16(val);
                } else if (mode == 2) {                          // (b,h,hd,s)
                    int bb = row >> 11, s = row & 2047;
                    int h  = col >> 6,  hd = col & 63;
                    outq[((size_t)(bb * 16 + h) * 64 + hd) * 2048 + s] =
                        f32_to_bf16(val);
                } else {                                         // fp32 row-major
                    outf[(size_t)row * N + col] = val;
                }
            }
        }
    }
}

// ---------------------------------------------------------------------------
// Flash attention: one block per (bh, 64-query tile), 128 threads = 4 waves.
// K in (b,h,s,hd), V pre-transposed to (b,h,hd,s): all tile staging is
// contiguous async-to-LDS, double-buffered across the key loop.
// ---------------------------------------------------------------------------
__global__ __launch_bounds__(128) void attn_fwd(const u16* __restrict__ Q,
                                                const u16* __restrict__ Kg,
                                                const u16* __restrict__ Vt,
                                                u16* __restrict__ Og) {
    __shared__ __align__(16) u16   Ks [2][64][64];   // [buf][key][hd]
    __shared__ __align__(16) u16   VT [2][64][64];   // [buf][hd][key]
    __shared__ __align__(16) float Sls[4][16][64];   // per-wave fp32 scores
    __shared__ __align__(16) u16   Pls[4][16][64];   // per-wave bf16 probs
    __shared__ float rowf[4][16];

    const int t    = threadIdx.x;
    const int lane = t & 31;
    const int wave = t >> 5;
    const int lh   = lane >> 4;
    const int lr   = lane & 15;
    const int qt   = blockIdx.x;
    const int bh   = blockIdx.y;
    const int q0   = qt * 64;
    const size_t base = (size_t)bh * 2048 * 64;      // same for K and Vt

    auto stageKV = [&](int buf, int kt) {
#pragma unroll
        for (int i = 0; i < 4; ++i) {
            int c   = t + 128 * i;             // 0..511
            int row = c >> 3;                  // key (for Ks) / hd (for VT)
            int ck  = (c & 7) << 3;
            cp16(&Ks[buf][row][ck], &Kg[base + (size_t)(kt * 64 + row) * 64 + ck]);
            cp16(&VT[buf][row][ck], &Vt[base + (size_t)row * 2048 + kt * 64 + ck]);
        }
    };

    // Q A-fragments (held in registers across the whole key loop)
    Frag qf[2];
    {
        int row = q0 + wave * 16 + lr;
        const u16* qp = &Q[base + (size_t)row * 64];
        qf[0].q[0] = *(const uint4*)&qp[lh * 8];
        qf[0].q[1] = *(const uint4*)&qp[16 + lh * 8];
        qf[1].q[0] = *(const uint4*)&qp[32 + lh * 8];
        qf[1].q[1] = *(const uint4*)&qp[48 + lh * 8];
    }

    v8f   o[4] = {};
    float m_run = -INFINITY;
    float l_run = 0.f;

    stageKV(0, 0);

    int buf = 0;
    for (int kt = 0; kt <= qt; ++kt, buf ^= 1) {
        async_wait();
        __syncthreads();                       // tiles[buf] ready; other buf free
        if (kt < qt) stageKV(buf ^ 1, kt + 1);

        // S = Q * K^T  (K^T B-fragments are contiguous in row-major Ks)
#pragma unroll
        for (int nt = 0; nt < 4; ++nt) {
            v8f sacc = {};
            int key = nt * 16 + lr;
#pragma unroll
            for (int ks = 0; ks < 2; ++ks) {
                Frag bf;
                bf.q[0] = *(const uint4*)&Ks[buf][key][ks * 32 + lh * 8];
                bf.q[1] = *(const uint4*)&Ks[buf][key][ks * 32 + 16 + lh * 8];
                sacc = __builtin_amdgcn_wmma_f32_16x16x32_bf16(
                    false, qf[ks].v, false, bf.v, (short)0, sacc, false, false);
            }
#pragma unroll
            for (int r = 0; r < 8; ++r)
                Sls[wave][r + 8 * lh][nt * 16 + lr] = sacc[r];
        }
        __syncthreads();

        // online softmax: lane < 16 owns query row (wave*16 + lane)
        if (lane < 16) {
            int lim = (kt == qt) ? (wave * 16 + lane + 1) : 64;   // causal limit
            float mt_ = -INFINITY;
            for (int c = 0; c < 64; ++c) {
                int cc = (c + lane) & 63;   // stagger lanes across banks
                if (cc < lim) {
                    float v = Sls[wave][lane][cc] * 0.125f;       // 1/sqrt(64)
                    mt_ = fmaxf(mt_, v);
                }
            }
            float newm = fmaxf(m_run, mt_);
            float fac  = __expf(m_run - newm);
            float sum  = 0.f;
            for (int c = 0; c < 64; ++c) {
                int cc = (c + lane) & 63;
                float p = 0.f;
                if (cc < lim)
                    p = __expf(Sls[wave][lane][cc] * 0.125f - newm);
                sum += p;
                Pls[wave][lane][cc] = f32_to_bf16(p);
            }
            l_run = l_run * fac + sum;
            m_run = newm;
            rowf[wave][lane] = fac;
        }
        __syncthreads();

        // rescale O accumulators by per-row factor
        float fr[8];
#pragma unroll
        for (int r = 0; r < 8; ++r) fr[r] = rowf[wave][r + 8 * lh];
#pragma unroll
        for (int nt = 0; nt < 4; ++nt)
#pragma unroll
            for (int r = 0; r < 8; ++r) o[nt][r] *= fr[r];

        // O += P * V  (V^T B-fragments contiguous per lane in VT)
#pragma unroll
        for (int ks = 0; ks < 2; ++ks) {
            Frag pf;
            pf.q[0] = *(const uint4*)&Pls[wave][lr][ks * 32 + lh * 8];
            pf.q[1] = *(const uint4*)&Pls[wave][lr][ks * 32 + 16 + lh * 8];
#pragma unroll
            for (int nt = 0; nt < 4; ++nt) {
                Frag vf;
                vf.q[0] = *(const uint4*)&VT[buf][nt * 16 + lr][ks * 32 + lh * 8];
                vf.q[1] = *(const uint4*)&VT[buf][nt * 16 + lr][ks * 32 + 16 + lh * 8];
                o[nt] = __builtin_amdgcn_wmma_f32_16x16x32_bf16(
                    false, pf.v, false, vf.v, (short)0, o[nt], false, false);
            }
        }
    }

    // finalize: divide by running sum, write bf16 O in (b*S+s, D) row-major
    __syncthreads();
    if (lane < 16) rowf[wave][lane] = 1.0f / l_run;
    __syncthreads();
    float inv[8];
#pragma unroll
    for (int r = 0; r < 8; ++r) inv[r] = rowf[wave][r + 8 * lh];

    const int b = bh >> 4, h = bh & 15;
#pragma unroll
    for (int nt = 0; nt < 4; ++nt) {
#pragma unroll
        for (int r = 0; r < 8; ++r) {
            int s  = q0 + wave * 16 + r + 8 * lh;
            int hd = nt * 16 + lr;
            float val = o[nt][r] * inv[r];
            Og[((size_t)(b * 2048 + s)) * 1024 + h * 64 + hd] = f32_to_bf16(val);
        }
    }
}

// ---------------------------------------------------------------------------
// launch
// ---------------------------------------------------------------------------
extern "C" void kernel_launch(void* const* d_in, const int* in_sizes, int n_in,
                              void* d_out, int out_size, void* d_ws, size_t ws_size,
                              hipStream_t stream) {
    const float* x  = (const float*)d_in[0];
    const float* Wq = (const float*)d_in[1];
    const float* bq = (const float*)d_in[2];
    const float* Wk = (const float*)d_in[3];
    const float* bk = (const float*)d_in[4];
    const float* Wv = (const float*)d_in[5];
    const float* bv = (const float*)d_in[6];
    const float* Wp = (const float*)d_in[7];
    const float* bp = (const float*)d_in[8];
    float* out = (float*)d_out;

    // workspace layout (bf16 buffers), all 256B-aligned by construction
    char* ws = (char*)d_ws;
    const size_t NX = (size_t)4 * 2048 * 1024;   // 8,388,608 elements
    const size_t NW = (size_t)1024 * 1024;       // 1,048,576 elements
    u16* xb  = (u16*)ws;                         // 16 MB
    u16* Wqt = xb  + NX;                         // transposed bf16 weights
    u16* Wkt = Wqt + NW;
    u16* Wvt = Wkt + NW;
    u16* Wpt = Wvt + NW;                         // +8 MB
    u16* Qb  = Wpt + NW;                         // 16 MB each below
    u16* Kb  = Qb  + NX;
    u16* Vb  = Kb  + NX;                         // V in (b,h,hd,s)
    u16* Ob  = Vb  + NX;                         // total ~88 MB

    // 1) conversions: x plain, weights transposed (one-time, so GEMM B-tiles
    //    and attention V-tiles are contiguous async-to-LDS copies)
    cvt_bf16<<<2048, 256, 0, stream>>>(x, xb, (int)NX);
    dim3 gt(16, 16);
    cvt_w_transpose<<<gt, 256, 0, stream>>>(Wq, Wqt);
    cvt_w_transpose<<<gt, 256, 0, stream>>>(Wk, Wkt);
    cvt_w_transpose<<<gt, 256, 0, stream>>>(Wv, Wvt);
    cvt_w_transpose<<<gt, 256, 0, stream>>>(Wp, Wpt);

    // 2) QKV projections: M=8192, N=1024
    dim3 gg(1024 / 128, 8192 / 128);             // (8, 64)
    gemm_bf16<<<gg, 256, 0, stream>>>(xb, Wqt, bq, Qb, nullptr, 0);
    gemm_bf16<<<gg, 256, 0, stream>>>(xb, Wkt, bk, Kb, nullptr, 0);
    gemm_bf16<<<gg, 256, 0, stream>>>(xb, Wvt, bv, Vb, nullptr, 2);  // V transposed

    // 3) causal flash attention -> Ob (b*S+s, D) bf16
    dim3 ga(2048 / 64, 4 * 16);                  // (32, 64)
    attn_fwd<<<ga, 128, 0, stream>>>(Qb, Kb, Vb, Ob);

    // 4) output projection -> fp32 d_out
    gemm_bf16<<<gg, 256, 0, stream>>>(Ob, Wpt, bp, nullptr, out, 1);
}
